// SpatialAttention_90520730730513
// MI455X (gfx1250) — matrix operand order repaired and checked by reading, not compiled, not run on hardware
//
#include <hip/hip_runtime.h>

typedef __attribute__((ext_vector_type(16))) __bf16 v16bf;
typedef __attribute__((ext_vector_type(8)))  __bf16 v8bf;
typedef __attribute__((ext_vector_type(4)))  __bf16 v4bf;
typedef __attribute__((ext_vector_type(8)))  float  v8f;

#define NNODES  512
#define HIDDEN  512
#define NEDGES  16384
#define HEADS   8
#define HDIM    64
#define NBATCH  48      // 4*12
#define NEG_INF (-1e9f)

#define SHUF16(lo, hi) __builtin_shufflevector(lo, hi, 0,1,2,3,4,5,6,7,8,9,10,11,12,13,14,15)

// ---- CDNA5 async DMA: global -> LDS, 16B per lane, tracked by ASYNCcnt ----
__device__ __forceinline__ void async_copy_b128(void* lds_dst, const void* gsrc) {
    unsigned lds = (unsigned)(size_t)lds_dst;
    asm volatile("global_load_async_to_lds_b128 %0, %1, off"
                 :: "v"(lds), "v"(gsrc) : "memory");
}
__device__ __forceinline__ void async_wait0() {
    asm volatile("s_wait_asynccnt 0x0" ::: "memory");
}

// ---------------- mask build ----------------
__global__ void k_mask_init(float* __restrict__ mask) {
    int i = blockIdx.x * blockDim.x + threadIdx.x;
    mask[i] = 0.0f;
}

__global__ void k_mask_scatter(const long long* __restrict__ ei, float* __restrict__ mask) {
    int i = blockIdx.x * blockDim.x + threadIdx.x;
    int r = (int)ei[i];
    int c = (int)ei[NEDGES + i];
    mask[r * NNODES + c] = 1.0f;
}

// ---------------- weight convert (keep [o][f] layout: it IS the B-fragment layout) ----------------
__global__ void k_cvt_w(const float* __restrict__ Wq, const float* __restrict__ Wk,
                        const float* __restrict__ Wv, __bf16* __restrict__ Wb) {
    int t = blockIdx.x * blockDim.x + threadIdx.x;
    int e = t * 4;
    int p = e >> 18;
    int rem = e & ((1 << 18) - 1);
    const float* W = (p == 0) ? Wq : ((p == 1) ? Wk : Wv);
    float4 f = *(const float4*)(W + rem);
    v4bf o = { (__bf16)f.x, (__bf16)f.y, (__bf16)f.z, (__bf16)f.w };
    *(v4bf*)(Wb + e) = o;
}

// ---------------- fused QKV projection: y = x @ W^T + b (double-buffered pipeline) ----------------
// Q stored [n][f]; K stored [n][f]; V stored transposed [f][n].
__launch_bounds__(256)
__global__ void k_qkv(const float* __restrict__ x, const __bf16* __restrict__ Wb,
                      const float* __restrict__ bq, const float* __restrict__ bk,
                      const float* __restrict__ bv,
                      __bf16* __restrict__ Qb, __bf16* __restrict__ Kb, __bf16* __restrict__ Vt) {
    __shared__ __align__(16) __bf16 As[2][64][40];   // [m-local][K 0..31], stride 80B
    __shared__ __align__(16) __bf16 Bs[2][64][40];   // [n-local][K 0..31], stride 80B

    const int t    = threadIdx.x;
    const int bs   = blockIdx.z / 3;
    const int proj = blockIdx.z % 3;
    const int m0   = blockIdx.y * 64;
    const int n0   = blockIdx.x * 64;
    const int w    = t >> 5;
    const int lane = t & 31;
    const int half = lane >> 4;
    const int l16  = lane & 15;
    const int mt   = w >> 1;               // 0..3
    const int ntb  = (w & 1) * 2;          // 0 or 2

    const float*  xb = x  + (size_t)bs   * HIDDEN * NNODES;
    const __bf16* Wp = Wb + (size_t)proj * HIDDEN * HIDDEN;

    // staging coords
    const int brow = t >> 2, bseg = (t & 3) * 8;     // B: 64 rows x 4 x 16B
    const int ar   = t >> 3, ac = (t & 7) * 4;       // A: rows ar, ar+32; 4 floats each

    v8f acc0 = {}; v8f acc1 = {};

    // ---- prologue: stage kt=0 into buffer 0 ----
    async_copy_b128(&Bs[0][brow][bseg], Wp + (size_t)(n0 + brow) * HIDDEN + bseg);
    {
        float4 f0 = *(const float4*)(xb + (size_t)(m0 + ar) * HIDDEN + ac);
        float4 f1 = *(const float4*)(xb + (size_t)(m0 + ar + 32) * HIDDEN + ac);
        v4bf o0v = { (__bf16)f0.x, (__bf16)f0.y, (__bf16)f0.z, (__bf16)f0.w };
        v4bf o1v = { (__bf16)f1.x, (__bf16)f1.y, (__bf16)f1.z, (__bf16)f1.w };
        *(v4bf*)&As[0][ar][ac]      = o0v;
        *(v4bf*)&As[0][ar + 32][ac] = o1v;
    }
    async_wait0();
    __syncthreads();

    int buf = 0;
    for (int kt = 0; kt < 16; ++kt) {
        const int  nb   = buf ^ 1;
        const bool more = (kt < 15);
        float4 f0{}, f1{};
        if (more) {   // issue next tile's DMA + global loads before compute
            const int k1 = (kt + 1) * 32;
            async_copy_b128(&Bs[nb][brow][bseg], Wp + (size_t)(n0 + brow) * HIDDEN + k1 + bseg);
            f0 = *(const float4*)(xb + (size_t)(m0 + ar) * HIDDEN + k1 + ac);
            f1 = *(const float4*)(xb + (size_t)(m0 + ar + 32) * HIDDEN + k1 + ac);
        }

        // ---- compute on current buffer ----
        const int arow = 16 * mt + l16;
        v8bf alo = *(const v8bf*)&As[buf][arow][16 * half];
        v8bf ahi = *(const v8bf*)&As[buf][arow][16 * half + 8];
        v16bf a  = SHUF16(alo, ahi);

        const int b0r = 16 * ntb + l16;
        v8bf b0lo = *(const v8bf*)&Bs[buf][b0r][16 * half];
        v8bf b0hi = *(const v8bf*)&Bs[buf][b0r][16 * half + 8];
        v16bf b0  = SHUF16(b0lo, b0hi);
        v8bf b1lo = *(const v8bf*)&Bs[buf][b0r + 16][16 * half];
        v8bf b1hi = *(const v8bf*)&Bs[buf][b0r + 16][16 * half + 8];
        v16bf b1  = SHUF16(b1lo, b1hi);

        acc0 = __builtin_amdgcn_wmma_f32_16x16x32_bf16(false, a, false, b0, (short)0, acc0, false, false);
        acc1 = __builtin_amdgcn_wmma_f32_16x16x32_bf16(false, a, false, b1, (short)0, acc1, false, false);

        if (more) {   // commit next A tile (loads have been in flight during compute)
            v4bf o0v = { (__bf16)f0.x, (__bf16)f0.y, (__bf16)f0.z, (__bf16)f0.w };
            v4bf o1v = { (__bf16)f1.x, (__bf16)f1.y, (__bf16)f1.z, (__bf16)f1.w };
            *(v4bf*)&As[nb][ar][ac]      = o0v;
            *(v4bf*)&As[nb][ar + 32][ac] = o1v;
            async_wait0();
        }
        __syncthreads();
        buf = nb;
    }

    const float* bias = (proj == 0) ? bq : ((proj == 1) ? bk : bv);
    __bf16* out = (proj == 0) ? Qb : ((proj == 1) ? Kb : Vt);
    out += (size_t)bs * HIDDEN * NNODES;

    #pragma unroll
    for (int r = 0; r < 8; ++r) {
        int gm  = m0 + 16 * mt + r + 8 * half;
        int gn0 = n0 + 16 * ntb + l16;
        int gn1 = n0 + 16 * (ntb + 1) + l16;
        float v0 = acc0[r] + bias[gn0];
        float v1 = acc1[r] + bias[gn1];
        if (proj == 2) {    // V stored transposed: Vt[f][n]
            out[gn0 * NNODES + gm] = (__bf16)v0;
            out[gn1 * NNODES + gm] = (__bf16)v1;
        } else {            // Q, K stored [n][f]
            out[gm * HIDDEN + gn0] = (__bf16)v0;
            out[gm * HIDDEN + gn1] = (__bf16)v1;
        }
    }
}

// ---------------- masked attention, flash-style (double-buffered K/V tiles) ----------------
__launch_bounds__(256)
__global__ void k_attn(const __bf16* __restrict__ Qb, const __bf16* __restrict__ Kb,
                       const __bf16* __restrict__ Vt, const float* __restrict__ mask,
                       float* __restrict__ out) {
    __shared__ __align__(16) __bf16 Kbs[2][32][72];   // [m-local][d 0..63], stride 144B
    __shared__ __align__(16) __bf16 Vts[2][64][40];   // [d][m-local 0..31], stride 80B
    __shared__ __align__(16) __bf16 Ps[8][16][40];    // per-wave P tile [M][K 0..31]

    const int t    = threadIdx.x;
    const int w    = t >> 5;
    const int lane = t & 31;
    const int half = lane >> 4;
    const int l16  = lane & 15;
    const int bh   = blockIdx.y;
    const int bs   = bh >> 3;
    const int h    = bh & 7;
    const int r0   = blockIdx.x * 128 + w * 16;
    const size_t base = (size_t)bs * HIDDEN * NNODES;
    const int h64  = h * HDIM;

    // Q A-fragments: 4x global b128 per lane, held in registers
    v16bf qa0, qa1;
    {
        const __bf16* qrow = Qb + base + (size_t)(r0 + l16) * HIDDEN + h64;
        v8bf q0lo = *(const v8bf*)(qrow + 16 * half);
        v8bf q0hi = *(const v8bf*)(qrow + 16 * half + 8);
        v8bf q1lo = *(const v8bf*)(qrow + 32 + 16 * half);
        v8bf q1hi = *(const v8bf*)(qrow + 40 + 16 * half);
        qa0 = SHUF16(q0lo, q0hi);
        qa1 = SHUF16(q1lo, q1hi);
    }

    v8f o0 = {}, o1 = {}, o2 = {}, o3 = {};
    float mrow[8], lrow[8];
    #pragma unroll
    for (int r = 0; r < 8; ++r) { mrow[r] = -3.0e38f; lrow[r] = 0.0f; }

    const int krow = t >> 3, kseg = (t & 7) * 8;   // K tile: 32 rows x 8 x 16B
    const int vrow = t >> 2, vseg = (t & 3) * 8;   // V tile: 64 rows x 4 x 16B

    // ---- prologue: stage chunk 0 into buffer 0 ----
    async_copy_b128(&Kbs[0][krow][kseg], Kb + base + (size_t)krow * HIDDEN + h64 + kseg);
    async_copy_b128(&Vts[0][vrow][vseg], Vt + base + (size_t)(h64 + vrow) * HIDDEN + vseg);
    async_wait0();
    __syncthreads();

    int buf = 0;
    for (int mc = 0; mc < 16; ++mc) {
        const int  m0   = mc * 32;
        const int  nb   = buf ^ 1;
        const bool more = (mc < 15);
        if (more) {   // issue next chunk's DMA before compute
            const int m1 = m0 + 32;
            async_copy_b128(&Kbs[nb][krow][kseg], Kb + base + (size_t)(m1 + krow) * HIDDEN + h64 + kseg);
            async_copy_b128(&Vts[nb][vrow][vseg], Vt + base + (size_t)(h64 + vrow) * HIDDEN + m1 + vseg);
        }

        // ---- S = Q K^T : two 16-col tiles, K(d)=64 = 2 wmma steps each ----
        v8f s0 = {}, s1 = {};
        #pragma unroll
        for (int j = 0; j < 2; ++j) {
            const int bro = 16 * j + l16;
            v8f s = {};
            v8bf blo = *(const v8bf*)&Kbs[buf][bro][16 * half];
            v8bf bhi = *(const v8bf*)&Kbs[buf][bro][16 * half + 8];
            v16bf b  = SHUF16(blo, bhi);
            s = __builtin_amdgcn_wmma_f32_16x16x32_bf16(false, qa0, false, b, (short)0, s, false, false);
            blo = *(const v8bf*)&Kbs[buf][bro][32 + 16 * half];
            bhi = *(const v8bf*)&Kbs[buf][bro][32 + 16 * half + 8];
            b   = SHUF16(blo, bhi);
            s = __builtin_amdgcn_wmma_f32_16x16x32_bf16(false, qa1, false, b, (short)0, s, false, false);
            if (j == 0) s0 = s; else s1 = s;
        }

        // ---- scale + graph mask ----
        #pragma unroll
        for (int r = 0; r < 8; ++r) {
            int row = r0 + r + 8 * half;
            float mk0 = mask[row * NNODES + m0 + l16];
            float mk1 = mask[row * NNODES + m0 + 16 + l16];
            s0[r] = (mk0 != 0.0f) ? s0[r] * 0.125f : NEG_INF;
            s1[r] = (mk1 != 0.0f) ? s1[r] * 0.125f : NEG_INF;
        }

        // ---- online softmax (16-lane half-group reductions) ----
        float alpha[8];
        #pragma unroll
        for (int r = 0; r < 8; ++r) {
            float mx = fmaxf(s0[r], s1[r]);
            #pragma unroll
            for (int d = 1; d < 16; d <<= 1) mx = fmaxf(mx, __shfl_xor(mx, d, 32));
            float mnew = fmaxf(mrow[r], mx);
            alpha[r] = __expf(mrow[r] - mnew);
            float p0 = __expf(s0[r] - mnew);
            float p1 = __expf(s1[r] - mnew);
            float ps = p0 + p1;
            #pragma unroll
            for (int d = 1; d < 16; d <<= 1) ps += __shfl_xor(ps, d, 32);
            lrow[r] = lrow[r] * alpha[r] + ps;
            mrow[r] = mnew;
            s0[r] = p0; s1[r] = p1;
        }

        // rescale accumulators; bounce P (C-layout -> A-layout) through LDS
        #pragma unroll
        for (int r = 0; r < 8; ++r) {
            o0[r] *= alpha[r]; o1[r] *= alpha[r]; o2[r] *= alpha[r]; o3[r] *= alpha[r];
            Ps[w][r + 8 * half][l16]      = (__bf16)s0[r];
            Ps[w][r + 8 * half][16 + l16] = (__bf16)s1[r];
        }
        v8bf plo = *(const v8bf*)&Ps[w][l16][16 * half];
        v8bf phi = *(const v8bf*)&Ps[w][l16][16 * half + 8];
        v16bf pa = SHUF16(plo, phi);

        // ---- O += P @ V over 4 d-tiles ----
        #pragma unroll
        for (int tt = 0; tt < 4; ++tt) {
            const int vro = 16 * tt + l16;
            v8bf blo = *(const v8bf*)&Vts[buf][vro][16 * half];
            v8bf bhi = *(const v8bf*)&Vts[buf][vro][16 * half + 8];
            v16bf vb = SHUF16(blo, bhi);
            if      (tt == 0) o0 = __builtin_amdgcn_wmma_f32_16x16x32_bf16(false, pa, false, vb, (short)0, o0, false, false);
            else if (tt == 1) o1 = __builtin_amdgcn_wmma_f32_16x16x32_bf16(false, pa, false, vb, (short)0, o1, false, false);
            else if (tt == 2) o2 = __builtin_amdgcn_wmma_f32_16x16x32_bf16(false, pa, false, vb, (short)0, o2, false, false);
            else              o3 = __builtin_amdgcn_wmma_f32_16x16x32_bf16(false, pa, false, vb, (short)0, o3, false, false);
        }

        if (more) async_wait0();   // DMA for next chunk has drained during compute
        __syncthreads();
        buf = nb;
    }

    // ---- normalize and store fp32 output ----
    #pragma unroll
    for (int r = 0; r < 8; ++r) {
        int row = r0 + r + 8 * half;
        float inv = 1.0f / lrow[r];
        size_t ob = base + (size_t)row * HIDDEN + h64;
        out[ob + l16]      = o0[r] * inv;
        out[ob + 16 + l16] = o1[r] * inv;
        out[ob + 32 + l16] = o2[r] * inv;
        out[ob + 48 + l16] = o3[r] * inv;
    }
}

// ---------------- host launcher ----------------
extern "C" void kernel_launch(void* const* d_in, const int* in_sizes, int n_in,
                              void* d_out, int out_size, void* d_ws, size_t ws_size,
                              hipStream_t stream) {
    const float*     x  = (const float*)d_in[0];
    const long long* ei = (const long long*)d_in[1];
    const float*     Wq = (const float*)d_in[2];
    const float*     bq = (const float*)d_in[3];
    const float*     Wk = (const float*)d_in[4];
    const float*     bk = (const float*)d_in[5];
    const float*     Wv = (const float*)d_in[6];
    const float*     bv = (const float*)d_in[7];
    float*           out = (float*)d_out;

    char* ws = (char*)d_ws;
    size_t o = 0;
    float*  mask = (float*)(ws + o);  o += (size_t)NNODES * NNODES * 4;
    __bf16* Wb   = (__bf16*)(ws + o); o += (size_t)3 * HIDDEN * HIDDEN * 2;
    __bf16* Qb   = (__bf16*)(ws + o); o += (size_t)NBATCH * NNODES * HIDDEN * 2;
    __bf16* Kb   = (__bf16*)(ws + o); o += (size_t)NBATCH * NNODES * HIDDEN * 2;
    __bf16* Vt   = (__bf16*)(ws + o); o += (size_t)NBATCH * NNODES * HIDDEN * 2;

    k_mask_init   <<<(NNODES * NNODES) / 256, 256, 0, stream>>>(mask);
    k_mask_scatter<<<NEDGES / 256, 256, 0, stream>>>(ei, mask);
    k_cvt_w       <<<(3 * HIDDEN * HIDDEN / 4) / 256, 256, 0, stream>>>(Wq, Wk, Wv, Wb);
    k_qkv         <<<dim3(8, 8, NBATCH * 3), 256, 0, stream>>>(x, Wb, bq, bk, bv, Qb, Kb, Vt);
    k_attn        <<<dim3(4, NBATCH * HEADS), 256, 0, stream>>>(Qb, Kb, Vt, mask, out);
}
